// PPAFormer_48095043781024
// MI455X (gfx1250) — compile-verified, matrix operand order checked by reading, not run on hardware
//
#include <hip/hip_runtime.h>
#include <hip/hip_bf16.h>
#include <math.h>

// ---------------------------------------------------------------------------
// PPAFormer block for MI455X (gfx1250). All 1x1 convs run as bf16 WMMA GEMMs
// (v_wmma_f32_16x16x32_bf16) fed by double-buffered async global->LDS DMA
// (global_load_async_to_lds_b128 + s_wait_asynccnt), with fused epilogues.
// LN/softmax/BN/dwconv are VALU kernels. Requires ws_size >= ~180 MB.
// ---------------------------------------------------------------------------

typedef __bf16 bf16;
typedef __attribute__((ext_vector_type(16))) __bf16 v16bf;
typedef __attribute__((ext_vector_type(8)))  __bf16 v8bf;
typedef __attribute__((ext_vector_type(8)))  float  v8f;

#define DEVI __device__ __forceinline__

static constexpr int BATCH = 4;
static constexpr int C     = 256;
static constexpr int HW    = 4096;     // 64*64
static constexpr int NPIX  = BATCH * HW;  // 16384
static constexpr int HID   = 1024;
static constexpr float BNRS = 0.99999500003749971f; // (1+1e-5)^-0.5

#define MB (1024ull * 1024ull)
// workspace arena (byte offsets); regions aliased by lifetime
static constexpr size_t O_TX    = 0;              // f32  [NPIX,256]   16MB
static constexpr size_t O_TXB   = 16 * MB;        // bf16 [NPIX,256]    8MB
static constexpr size_t O_N1    = 24 * MB;        // bf16 [NPIX,256]    8MB
static constexpr size_t O_QKV   = 32 * MB;        // bf16 [NPIX,768]   24MB
static constexpr size_t O_ATT   = 56 * MB;        // bf16 [NPIX,256]    8MB
static constexpr size_t O_CAT   = 64 * MB;        // bf16 [NPIX,1280]  40MB
static constexpr size_t O_H     = 32 * MB;        // f32  [NPIX,1024]  64MB (aliases QKV/ATT/CAT after they die)
static constexpr size_t O_TR    = 104 * MB;       // f32  [NPIX,256]   16MB
static constexpr size_t O_LN2   = 120 * MB;       // bf16 [NPIX,256]    8MB
static constexpr size_t O_H2    = 128 * MB;       // bf16 [NPIX,1024]  32MB
static constexpr size_t O_OUTB  = 160 * MB;       // f32  [NPIX,256]   16MB
static constexpr size_t O_POOLF = 176 * MB;       // f32  [4,256]
static constexpr size_t O_POOLB = 176 * MB + 8192;// bf16 [4,256]
static constexpr size_t O_WB    = 176 * MB + 16384;// bf16 weight arena (~3.4MB)

// element offsets inside bf16 weight arena
static constexpr size_t WOFF_QKV   = 0;                       // 3 x 768*256
static constexpr size_t WOFF_PROJ  = WOFF_QKV  + 3 * 196608;  // 3 x 256*256
static constexpr size_t WOFF_TCONV = WOFF_PROJ + 3 * 65536;   // 256*256
static constexpr size_t WOFF_TRANS = WOFF_TCONV + 65536;      // 256*1280
static constexpr size_t WOFF_C1    = WOFF_TRANS + 327680;     // 1024*256
static constexpr size_t WOFF_C2    = WOFF_C1    + 262144;     // 256*1024

DEVI float b2f_lo(unsigned u) { union { unsigned x; float f; } c; c.x = u << 16; return c.f; }
DEVI float b2f_hi(unsigned u) { union { unsigned x; float f; } c; c.x = u & 0xffff0000u; return c.f; }
DEVI unsigned short f2b(float f) {
  union { float f; unsigned u; } c; c.f = f;
  unsigned u = c.u;
  return (unsigned short)((u + 0x7fffu + ((u >> 16) & 1u)) >> 16);   // RNE
}
DEVI float gelu_exact(float x) { return 0.5f * x * (1.0f + erff(x * 0.70710678118654752f)); }

// ---------------------------------------------------------------------------
// gfx1250 async global->LDS copy: per-lane 16B, one ASYNCcnt per instruction.
// VDST = LDS byte address (low 32 bits of flat shared addr), VADDR = global.
// ---------------------------------------------------------------------------
DEVI void async_copy_b128(const bf16* g, bf16* lds_generic) {
  unsigned loff = (unsigned)(uintptr_t)lds_generic;   // low 32 bits = LDS byte addr
  asm volatile("global_load_async_to_lds_b128 %0, %1, off"
               :: "v"(loff), "v"(g)
               : "memory");
}
template <int N>
DEVI void wait_async() {
  asm volatile("s_wait_asynccnt %0" :: "i"(N) : "memory");
}

// ---------------------------------------------------------------------------
// WMMA fragment load: 16-bit A/B fragment, 16x32 (rows x K), from row-major
// [rows, K=ld] bf16 memory (global or LDS). ISA layout: lanes 0-15 row=lane,
// K=0..7 (+16..23 in upper 4 VGPRs); lanes 16-31 same rows, K=8..15 (+24..31).
// ---------------------------------------------------------------------------
DEVI v16bf load_frag(const bf16* __restrict__ base, int ld, int lane) {
  int r  = lane & 15;
  int hi = (lane >> 4) & 1;
  const bf16* p = base + (size_t)r * ld + hi * 8;
  union { v16bf v; v8bf h[2]; } u;
  u.h[0] = *reinterpret_cast<const v8bf*>(p);
  u.h[1] = *reinterpret_cast<const v8bf*>(p + 16);
  return u.v;
}

// epilogue kinds
enum {
  E_STORE_BF = 0,        // qkv: plain bf16 store
  E_BIAS_RES_BF,         // proj: +bias +resid(f32) -> bf16 (into cat slice)
  E_BIAS_BN_RELU_BF,     // xc1: bn(x+bias) relu -> bf16 (into cat slice)
  E_BIAS_BN_RELU_F,      // trans: bn(x+bias) relu -> f32 (tr)
  E_BIAS_GELU_BN_F,      // mlp c1: bn(gelu(x+bias)) -> f32 (h)
  E_BIAS_BN_RES_F        // mlp c2: bn(x+bias)+resid -> f32 (out bhwc)
};

// D = A[M,K] * Bw^T, Bw row-major [Cout,K] (== reference 'oc' weights).
// Block: 256 threads = 8 waves; block tile 64(M) x 256(N); wave tile 32x64.
// A/B K-step tiles double-buffered in LDS via async DMA.
template <int EPI>
__global__ __launch_bounds__(256) void gemm_wmma(
    const bf16* __restrict__ A, const bf16* __restrict__ Bw, int K, int ldA,
    bf16* __restrict__ outB, float* __restrict__ outF, int ldOut, int chOff,
    const float* __restrict__ bias, const float* __restrict__ bng,
    const float* __restrict__ bnb, const float* __restrict__ resid, int ldRes) {
  __shared__ bf16 As[2][64 * 32];    // 4KB per stage
  __shared__ bf16 Bs[2][256 * 32];   // 16KB per stage

  const int tid  = threadIdx.x;
  const int lane = tid & 31;
  const int wv   = tid >> 5;
  const int wy   = wv & 1;        // M split (2)
  const int wx   = wv >> 1;       // N split (4)
  const int mB   = blockIdx.x * 64;
  const int nB   = blockIdx.y * 256;

  // stage prefetch: A tile 64x32 (1 async op/wave), B tile 256x32 (4 ops/wave)
  auto prefetch = [&](int k0, int buf) {
    {
      const int row = tid >> 2, sub = tid & 3;           // 256 chunks of 16B
      async_copy_b128(A + (size_t)(mB + row) * ldA + k0 + sub * 8,
                      &As[buf][row * 32 + sub * 8]);
    }
#pragma unroll
    for (int j = 0; j < 4; j++) {
      const int ch = tid + 256 * j;                      // 1024 chunks of 16B
      const int row = ch >> 2, sub = ch & 3;
      async_copy_b128(Bw + (size_t)(nB + row) * K + k0 + sub * 8,
                      &Bs[buf][row * 32 + sub * 8]);
    }
  };

  v8f acc[2][4];
#pragma unroll
  for (int i = 0; i < 2; i++)
#pragma unroll
    for (int j = 0; j < 4; j++)
#pragma unroll
      for (int r = 0; r < 8; r++) acc[i][j][r] = 0.0f;

  const int nk = K >> 5;
  prefetch(0, 0);
  for (int ks = 0; ks < nk; ks++) {
    const int buf = ks & 1;
    if (ks + 1 < nk) {
      prefetch((ks + 1) << 5, buf ^ 1);  // overlap next-stage DMA with math
      wait_async<5>();                   // this stage's 5 ops done (in-order)
    } else {
      wait_async<0>();
    }
    __syncthreads();

    v16bf af[2], bfr[4];
#pragma unroll
    for (int i = 0; i < 2; i++)
      af[i] = load_frag(&As[buf][(wy * 32 + i * 16) * 32], 32, lane);
#pragma unroll
    for (int j = 0; j < 4; j++)
      bfr[j] = load_frag(&Bs[buf][(wx * 64 + j * 16) * 32], 32, lane);
#pragma unroll
    for (int i = 0; i < 2; i++)
#pragma unroll
      for (int j = 0; j < 4; j++)
        acc[i][j] = __builtin_amdgcn_wmma_f32_16x16x32_bf16(
            false, af[i], false, bfr[j], (short)0, acc[i][j], false, false);
    __syncthreads();   // all reads of `buf` done before it is re-filled
  }

  // D layout: lanes 0-15 col n = lane, row = vgpr r; lanes 16-31 row = r+8
  const int rlo = lane & 15;
  const int hi  = lane >> 4;
  const int m0  = mB + wy * 32;
  const int n0  = nB + wx * 64;
#pragma unroll
  for (int i = 0; i < 2; i++)
#pragma unroll
    for (int j = 0; j < 4; j++)
#pragma unroll
      for (int r = 0; r < 8; r++) {
        const int m = m0 + i * 16 + hi * 8 + r;
        const int n = n0 + j * 16 + rlo;
        float v = acc[i][j][r];
        if (EPI == E_STORE_BF) {
          ((unsigned short*)outB)[(size_t)m * ldOut + chOff + n] = f2b(v);
        } else if (EPI == E_BIAS_RES_BF) {
          v += bias[n] + resid[(size_t)m * ldRes + n];
          ((unsigned short*)outB)[(size_t)m * ldOut + chOff + n] = f2b(v);
        } else if (EPI == E_BIAS_BN_RELU_BF) {
          v = (v + bias[n]) * (bng[n] * BNRS) + bnb[n];
          v = fmaxf(v, 0.0f);
          ((unsigned short*)outB)[(size_t)m * ldOut + chOff + n] = f2b(v);
        } else if (EPI == E_BIAS_BN_RELU_F) {
          v = (v + bias[n]) * (bng[n] * BNRS) + bnb[n];
          outF[(size_t)m * ldOut + chOff + n] = fmaxf(v, 0.0f);
        } else if (EPI == E_BIAS_GELU_BN_F) {
          v = gelu_exact(v + bias[n]) * (bng[n] * BNRS) + bnb[n];
          outF[(size_t)m * ldOut + chOff + n] = v;
        } else { // E_BIAS_BN_RES_F
          v = (v + bias[n]) * (bng[n] * BNRS) + bnb[n] + resid[(size_t)m * ldRes + n];
          outF[(size_t)m * ldOut + chOff + n] = v;
        }
      }
}

// ---------------------------------------------------------------------------
// NCHW -> pixel-major transpose (f32 + bf16 copies). grid (128,8,4), blk(32,8)
// ---------------------------------------------------------------------------
__global__ void k_transpose_in(const float* __restrict__ x, float* __restrict__ tx,
                               bf16* __restrict__ txb) {
  __shared__ float t[32][33];
  const int b = blockIdx.z, hw0 = blockIdx.x * 32, c0 = blockIdx.y * 32;
  const int lx = threadIdx.x, ly = threadIdx.y;
  for (int i = ly; i < 32; i += 8)
    t[i][lx] = x[(size_t)(b * C + c0 + i) * HW + hw0 + lx];
  __syncthreads();
  for (int i = ly; i < 32; i += 8) {
    const float v = t[lx][i];   // c = c0+lx, hw = hw0+i
    const size_t o = (size_t)(b * HW + hw0 + i) * C + c0 + lx;
    tx[o] = v;
    ((unsigned short*)txb)[o] = f2b(v);
  }
}

// pixel-major f32 -> NCHW output
__global__ void k_transpose_out(const float* __restrict__ src, float* __restrict__ dst) {
  __shared__ float t[32][33];
  const int b = blockIdx.z, hw0 = blockIdx.x * 32, c0 = blockIdx.y * 32;
  const int lx = threadIdx.x, ly = threadIdx.y;
  for (int i = ly; i < 32; i += 8)
    t[i][lx] = src[(size_t)(b * HW + hw0 + i) * C + c0 + lx]; // t[hw][c]
  __syncthreads();
  for (int i = ly; i < 32; i += 8)
    dst[(size_t)(b * C + c0 + i) * HW + hw0 + lx] = t[lx][i];
}

// LayerNorm over C=256; one wave per pixel, 4 pixels per block
__global__ __launch_bounds__(128) void k_ln(const float* __restrict__ in,
                                            bf16* __restrict__ out,
                                            const float* __restrict__ w,
                                            const float* __restrict__ b) {
  const int lane = threadIdx.x & 31;
  const int pix  = blockIdx.x * 4 + (threadIdx.x >> 5);
  const float* row = in + (size_t)pix * C;
  float vals[8];
#pragma unroll
  for (int i = 0; i < 8; i++) vals[i] = row[lane * 8 + i];
  float s = 0.f, q = 0.f;
#pragma unroll
  for (int i = 0; i < 8; i++) { s += vals[i]; q += vals[i] * vals[i]; }
  for (int o = 16; o > 0; o >>= 1) { s += __shfl_xor(s, o, 32); q += __shfl_xor(q, o, 32); }
  const float mean = s * (1.0f / 256.0f);
  const float var  = q * (1.0f / 256.0f) - mean * mean;
  const float rstd = rsqrtf(var + 1e-5f);
  unsigned short* op = (unsigned short*)out + (size_t)pix * C + lane * 8;
#pragma unroll
  for (int i = 0; i < 8; i++) {
    const int c = lane * 8 + i;
    op[i] = f2b((vals[i] - mean) * rstd * w[c] + b[c]);
  }
}

// per-(b,c) spatial mean of NCHW x
__global__ __launch_bounds__(256) void k_pool_reduce(const float* __restrict__ x,
                                                     float* __restrict__ poolf) {
  const int bc = blockIdx.x;               // 0..1023
  const float* p = x + (size_t)bc * HW;
  float s = 0.f;
  for (int i = threadIdx.x; i < HW; i += 256) s += p[i];
  __shared__ float sm[256];
  sm[threadIdx.x] = s;
  __syncthreads();
  for (int o = 128; o > 0; o >>= 1) {
    if (threadIdx.x < o) sm[threadIdx.x] += sm[threadIdx.x + o];
    __syncthreads();
  }
  if (threadIdx.x == 0) poolf[bc] = sm[0] * (1.0f / HW);
}

// pool @ tc_pool_w^T, bn2, relu -> bf16 [4,256]
__global__ __launch_bounds__(256) void k_pool_mv(const float* __restrict__ poolf,
                                                 const float* __restrict__ w,
                                                 const float* __restrict__ g,
                                                 const float* __restrict__ bb,
                                                 bf16* __restrict__ poolb) {
  const int b = blockIdx.x, o = threadIdx.x;
  const float* pv = poolf + b * C;
  const float* wr = w + (size_t)o * C;
  float s = 0.f;
  for (int c = 0; c < C; c++) s += pv[c] * wr[c];
  s = s * (g[o] * BNRS) + bb[o];
  ((unsigned short*)poolb)[b * C + o] = f2b(fmaxf(s, 0.0f));
}

// broadcast pool vector into cat channels [4C..5C)
__global__ void k_fill_pool(const bf16* __restrict__ poolb, bf16* __restrict__ cat) {
  const int idx = blockIdx.x * 256 + threadIdx.x; // NPIX*256 total
  const int pix = idx >> 8, c = idx & 255;
  ((unsigned short*)cat)[(size_t)pix * 1280 + 1024 + c] =
      ((const unsigned short*)poolb)[(pix >> 12) * C + c];
}

// atrous 9-neighbor attention; qkv bf16 [NPIX,768] (q|k|v), out bf16 [NPIX,256]
// OOB neighbors contribute logit 0 (zero-padded k) and v=0, matching unfold.
__global__ __launch_bounds__(256) void k_attn(const bf16* __restrict__ qkv,
                                              bf16* __restrict__ out, int r) {
  const int t = blockIdx.x * 256 + threadIdx.x;  // NPIX*8
  const int pix = t >> 3, head = t & 7;
  const int b = pix >> 12, hw = pix & 4095, h = hw >> 6, w = hw & 63;
  const unsigned* qp = (const unsigned*)qkv + ((size_t)pix * 768 + head * 32) / 2;
  float q[32];
#pragma unroll
  for (int i = 0; i < 16; i++) {
    const unsigned u = qp[i];
    q[2 * i] = b2f_lo(u); q[2 * i + 1] = b2f_hi(u);
  }
  const float scale = 0.17677669529663687f; // 1/sqrt(32)
  float lg[9];
  int np[9];
  bool val[9];
#pragma unroll
  for (int j = 0; j < 9; j++) {
    const int dy = (j / 3 - 1) * r, dx = (j % 3 - 1) * r;
    const int hh = h + dy, ww = w + dx;
    const bool v = (hh >= 0) && (hh < 64) && (ww >= 0) && (ww < 64);
    val[j] = v;
    np[j] = (b << 12) | ((hh & 63) << 6) | (ww & 63);
    float d = 0.f;
    if (v) {
      const unsigned* kp = (const unsigned*)qkv + ((size_t)np[j] * 768 + 256 + head * 32) / 2;
#pragma unroll
      for (int i = 0; i < 16; i++) {
        const unsigned u = kp[i];
        d += q[2 * i] * b2f_lo(u) + q[2 * i + 1] * b2f_hi(u);
      }
    }
    lg[j] = v ? d * scale : 0.0f;
  }
  float mx = lg[0];
#pragma unroll
  for (int j = 1; j < 9; j++) mx = fmaxf(mx, lg[j]);
  float e[9], sum = 0.f;
#pragma unroll
  for (int j = 0; j < 9; j++) { e[j] = expf(lg[j] - mx); sum += e[j]; }
  const float inv = 1.0f / sum;
  float o[32];
#pragma unroll
  for (int i = 0; i < 32; i++) o[i] = 0.f;
#pragma unroll
  for (int j = 0; j < 9; j++) {
    if (!val[j]) continue;
    const float p = e[j] * inv;
    const unsigned* vp = (const unsigned*)qkv + ((size_t)np[j] * 768 + 512 + head * 32) / 2;
#pragma unroll
    for (int i = 0; i < 16; i++) {
      const unsigned u = vp[i];
      o[2 * i]     += p * b2f_lo(u);
      o[2 * i + 1] += p * b2f_hi(u);
    }
  }
  unsigned* op = (unsigned*)out + ((size_t)pix * C + head * 32) / 2;
#pragma unroll
  for (int i = 0; i < 16; i++)
    op[i] = (unsigned)f2b(o[2 * i]) | ((unsigned)f2b(o[2 * i + 1]) << 16);
}

// depthwise 3x3 + bias + residual(h) -> gelu -> bn2 -> bf16
__global__ __launch_bounds__(256) void k_dw(const float* __restrict__ hbuf,
                                            const float* __restrict__ w,
                                            const float* __restrict__ bias,
                                            const float* __restrict__ g,
                                            const float* __restrict__ bb,
                                            bf16* __restrict__ out) {
  const size_t idx = (size_t)blockIdx.x * 256 + threadIdx.x; // NPIX*1024
  const int c = (int)(idx & 1023);
  const int pix = (int)(idx >> 10);
  const int b = pix >> 12, hw = pix & 4095, h = hw >> 6, ww = hw & 63;
  float acc = 0.f;
#pragma unroll
  for (int ky = 0; ky < 3; ky++)
#pragma unroll
    for (int kx = 0; kx < 3; kx++) {
      const int hh = h + ky - 1, wx = ww + kx - 1;
      if (hh >= 0 && hh < 64 && wx >= 0 && wx < 64) {
        const int npix = (b << 12) | (hh << 6) | wx;
        acc += w[c * 9 + ky * 3 + kx] * hbuf[(size_t)npix * HID + c];
      }
    }
  const float center = hbuf[(size_t)pix * HID + c];
  float v = gelu_exact(acc + bias[c] + center);
  v = v * (g[c] * BNRS) + bb[c];
  ((unsigned short*)out)[(size_t)pix * HID + c] = f2b(v);
}

// f32 -> bf16 weight conversion
__global__ void k_f2b(const float* __restrict__ in, bf16* __restrict__ out, int n) {
  const int i = blockIdx.x * 256 + threadIdx.x;
  if (i < n) ((unsigned short*)out)[i] = f2b(in[i]);
}

// ---------------------------------------------------------------------------
extern "C" void kernel_launch(void* const* d_in, const int* in_sizes, int n_in,
                              void* d_out, int out_size, void* d_ws, size_t ws_size,
                              hipStream_t stream) {
  (void)in_sizes; (void)n_in; (void)out_size; (void)ws_size;
  char* ws = (char*)d_ws;

  const float* x      = (const float*)d_in[0];
  const float* ln1_w  = (const float*)d_in[1];
  const float* ln1_b  = (const float*)d_in[2];
  const float* ln2_w  = (const float*)d_in[3];
  const float* ln2_b  = (const float*)d_in[4];
  const float* qkv_w[3]  = { (const float*)d_in[5], (const float*)d_in[8],  (const float*)d_in[11] };
  const float* proj_w[3] = { (const float*)d_in[6], (const float*)d_in[9],  (const float*)d_in[12] };
  const float* proj_b[3] = { (const float*)d_in[7], (const float*)d_in[10], (const float*)d_in[13] };
  const float* tc_conv_w = (const float*)d_in[14];
  const float* tc_conv_b = (const float*)d_in[15];
  const float* tc_bn1_g  = (const float*)d_in[16];
  const float* tc_bn1_b  = (const float*)d_in[17];
  const float* tc_pool_w = (const float*)d_in[18];
  const float* tc_bn2_g  = (const float*)d_in[19];
  const float* tc_bn2_b  = (const float*)d_in[20];
  const float* tc_trans_w= (const float*)d_in[21];
  const float* tc_trans_b= (const float*)d_in[22];
  const float* tc_bn3_g  = (const float*)d_in[23];
  const float* tc_bn3_b  = (const float*)d_in[24];
  const float* c1_w  = (const float*)d_in[25];
  const float* c1_b  = (const float*)d_in[26];
  const float* bn1_g = (const float*)d_in[27];
  const float* bn1_b = (const float*)d_in[28];
  const float* dw_w  = (const float*)d_in[29];
  const float* dw_b  = (const float*)d_in[30];
  const float* bn2_g = (const float*)d_in[31];
  const float* bn2_b = (const float*)d_in[32];
  const float* c2_w  = (const float*)d_in[33];
  const float* c2_b  = (const float*)d_in[34];
  const float* bn3_g = (const float*)d_in[35];
  const float* bn3_b = (const float*)d_in[36];

  float* tx   = (float*)(ws + O_TX);
  bf16*  txb  = (bf16*)(ws + O_TXB);
  bf16*  n1   = (bf16*)(ws + O_N1);
  bf16*  qkvb = (bf16*)(ws + O_QKV);
  bf16*  attb = (bf16*)(ws + O_ATT);
  bf16*  catb = (bf16*)(ws + O_CAT);
  float* hbuf = (float*)(ws + O_H);
  float* tr   = (float*)(ws + O_TR);
  bf16*  ln2b = (bf16*)(ws + O_LN2);
  bf16*  h2   = (bf16*)(ws + O_H2);
  float* obhwc= (float*)(ws + O_OUTB);
  float* poolf= (float*)(ws + O_POOLF);
  bf16*  poolb= (bf16*)(ws + O_POOLB);
  bf16*  warena = (bf16*)(ws + O_WB);

  // weight conversions f32 -> bf16
  auto conv = [&](const float* src, bf16* dst, int n) {
    k_f2b<<<(n + 255) / 256, 256, 0, stream>>>(src, dst, n);
  };
  for (int i = 0; i < 3; i++) {
    conv(qkv_w[i],  warena + WOFF_QKV  + (size_t)i * 196608, 768 * 256);
    conv(proj_w[i], warena + WOFF_PROJ + (size_t)i * 65536,  256 * 256);
  }
  conv(tc_conv_w,  warena + WOFF_TCONV, 256 * 256);
  conv(tc_trans_w, warena + WOFF_TRANS, 256 * 1280);
  conv(c1_w,       warena + WOFF_C1,    1024 * 256);
  conv(c2_w,       warena + WOFF_C2,    256 * 1024);

  // stage 1: layout + LN1 + pool branch
  k_transpose_in<<<dim3(128, 8, 4), dim3(32, 8), 0, stream>>>(x, tx, txb);
  k_ln<<<NPIX / 4, 128, 0, stream>>>(tx, n1, ln1_w, ln1_b);
  k_pool_reduce<<<BATCH * C, 256, 0, stream>>>(x, poolf);
  k_pool_mv<<<BATCH, 256, 0, stream>>>(poolf, tc_pool_w, tc_bn2_g, tc_bn2_b, poolb);
  k_fill_pool<<<NPIX, 256, 0, stream>>>(poolb, catb);

  // xc1 = relu(bn1(tx @ tc_conv_w^T + b)) -> cat[:, 0:256]
  gemm_wmma<E_BIAS_BN_RELU_BF><<<dim3(NPIX / 64, 1), 256, 0, stream>>>(
      txb, warena + WOFF_TCONV, 256, 256, catb, nullptr, 1280, 0,
      tc_conv_b, tc_bn1_g, tc_bn1_b, nullptr, 0);

  // three atrous attentions -> cat[:, 256*r : 256*(r+1)]
  for (int i = 0; i < 3; i++) {
    gemm_wmma<E_STORE_BF><<<dim3(NPIX / 64, 3), 256, 0, stream>>>(
        n1, warena + WOFF_QKV + (size_t)i * 196608, 256, 256, qkvb, nullptr, 768, 0,
        nullptr, nullptr, nullptr, nullptr, 0);
    k_attn<<<NPIX * 8 / 256, 256, 0, stream>>>(qkvb, attb, i + 1);
    gemm_wmma<E_BIAS_RES_BF><<<dim3(NPIX / 64, 1), 256, 0, stream>>>(
        attb, warena + WOFF_PROJ + (size_t)i * 65536, 256, 256, catb, nullptr,
        1280, 256 * (i + 1), proj_b[i], nullptr, nullptr, tx, 256);
  }

  // trans: relu(bn3(cat @ tc_trans_w^T + b)) -> tr (f32)
  gemm_wmma<E_BIAS_BN_RELU_F><<<dim3(NPIX / 64, 1), 256, 0, stream>>>(
      catb, warena + WOFF_TRANS, 1280, 1280, nullptr, tr, 256, 0,
      tc_trans_b, tc_bn3_g, tc_bn3_b, nullptr, 0);

  // MLP
  k_ln<<<NPIX / 4, 128, 0, stream>>>(tr, ln2b, ln2_w, ln2_b);
  gemm_wmma<E_BIAS_GELU_BN_F><<<dim3(NPIX / 64, 4), 256, 0, stream>>>(
      ln2b, warena + WOFF_C1, 256, 256, nullptr, hbuf, 1024, 0,
      c1_b, bn1_g, bn1_b, nullptr, 0);
  k_dw<<<(int)((size_t)NPIX * HID / 256), 256, 0, stream>>>(
      hbuf, dw_w, dw_b, bn2_g, bn2_b, h2);
  gemm_wmma<E_BIAS_BN_RES_F><<<dim3(NPIX / 64, 1), 256, 0, stream>>>(
      h2, warena + WOFF_C2, 1024, 1024, nullptr, obhwc, 256, 0,
      c2_b, bn3_g, bn3_b, tr, 256);

  // back to NCHW
  k_transpose_out<<<dim3(128, 8, 4), dim3(32, 8), 0, stream>>>(obhwc, (float*)d_out);
}
